// Attention_Layer_76098230550576
// MI455X (gfx1250) — matrix operand order repaired and checked by reading, not compiled, not run on hardware
//
#include <hip/hip_runtime.h>
#include <hip/hip_bf16.h>
#include <math.h>

#define HIDC 256
#define POSD 32
#define NHEADS 4
#define DHEAD 64
#define BATCH 8
#define SEQN 2048
#define ROWS (BATCH * SEQN)      // 16384
#define CIN (HIDC + POSD)        // 288
#define KBLK 64                  // keys per staged block
#define LN_EPS 1e-5f

#if __has_builtin(__builtin_amdgcn_tensor_load_to_lds) && \
    __has_builtin(__builtin_amdgcn_s_wait_tensorcnt)
#define HAVE_TDM 1
#else
#define HAVE_TDM 0
#endif

typedef __attribute__((ext_vector_type(16))) __bf16 v16bf;
typedef __attribute__((ext_vector_type(8)))  __bf16 v8bf;
typedef __attribute__((ext_vector_type(8)))  float  v8f;

union AFrag { v16bf v; v8bf h[2]; };
union BFrag { v16bf v; v8bf h[2]; };

__device__ __forceinline__ v8f wmma_bf16(v16bf a, v16bf b, v8f c) {
    return __builtin_amdgcn_wmma_f32_16x16x32_bf16(false, a, false, b, (short)0, c,
                                                   false, false);
}

__device__ __forceinline__ __bf16 tobf(float f) { return (__bf16)f; }

// XOR-swizzle of 16B groups within a 64-element (128B) row (fallback path).
__device__ __forceinline__ int swz(int row, int col) {
    int g = col >> 3, in = col & 7;
    return (((g ^ (row & 7)) << 3) | in);
}

#if HAVE_TDM
typedef __attribute__((ext_vector_type(4))) unsigned int u32x4;
typedef __attribute__((ext_vector_type(8))) int i32x8;
typedef __attribute__((ext_vector_type(4))) int i32x4;

// One-instruction DMA of a 64x64 bf16 tile (rows strided by rowStrideElems in
// global memory) into LDS, with 32B LDS padding per 128B row (bank de-conflict).
// D# layout per CDNA5 ISA ch.8: group0 = {count, lds_addr, global_addr, type},
// group1 = {flags/data_size/pad, tensor dims, tile dims, dim0 stride}.
// This toolchain exposes the 6-arg builtin: (g0, g1, g2, g3, extra8, cpol).
__device__ __forceinline__ void tdm_load_2d(const __bf16* gsrc, const __bf16* ldst,
                                            int rowStrideElems) {
    unsigned long long ga = (unsigned long long)(uintptr_t)gsrc;
    unsigned lds = (unsigned)(uintptr_t)ldst;   // low 32 bits = LDS byte offset
    u32x4 g0;
    g0[0] = 1u;                                          // count=1 (valid D#)
    g0[1] = lds;                                         // lds_addr
    g0[2] = (unsigned)ga;                                // global_addr[31:0]
    g0[3] = (unsigned)((ga >> 32) & 0x01FFFFFFu)         // global_addr[56:32]
            | 0x80000000u;                               // type=2 ("image")
    i32x8 g1;
    g1[0] = (int)((1u << 16)      // data_size = 2 bytes
                | (1u << 20)      // pad_enable
                | (4u << 22)      // pad_interval: every 32 DWORDs (=128B row)
                | (7u << 25));    // pad_amount: 8 DWORDs (=32B)
    g1[1] = (int)(64u << 16);     // tensor_dim0 = 64  (bits [79:48] low half)
    g1[2] = (int)(64u << 16);     // tensor_dim1 = 64  (bits [111:80] low half)
    g1[3] = (int)(64u << 16);     // tile_dim0  = 64  (bits [127:112])
    g1[4] = 64;                   // tile_dim1  = 64  (bits [143:128])
    g1[5] = rowStrideElems;       // tensor_dim0_stride (bits [207:160] low)
    g1[6] = 0;
    g1[7] = 0;
    i32x4 gz4 = {0, 0, 0, 0};
    i32x8 gz8 = {0, 0, 0, 0, 0, 0, 0, 0};
    __builtin_amdgcn_tensor_load_to_lds(g0, g1, gz4, gz4, gz8, 0);
}
#define KSTR 80                    // padded LDS row stride (64 + 16 elems)
#else
#define KSTR 64
#endif

// ---------------------------------------------------------------------------
// 1) Fold in_proj into Wq/Wk/Wv:  eff[kind][n,c] = sum_j ipw[kind*256+n, j]*W[j, c]
// ---------------------------------------------------------------------------
__global__ void prep_eff_kernel(const float* __restrict__ ipw,
                                const float* __restrict__ Wq,
                                const float* __restrict__ Wk,
                                const float* __restrict__ Wv,
                                __bf16* __restrict__ eff) {
    int idx = blockIdx.x * blockDim.x + threadIdx.x;
    int kind = blockIdx.y;
    if (idx >= HIDC * CIN) return;
    int n = idx / CIN, c = idx % CIN;
    const float* W = (kind == 0) ? Wq : (kind == 1) ? Wk : Wv;
    const float* wrow = ipw + (size_t)(kind * HIDC + n) * HIDC;
    float s = 0.f;
    for (int j = 0; j < HIDC; ++j) s += wrow[j] * W[(size_t)j * CIN + c];
    eff[(size_t)kind * HIDC * CIN + (size_t)n * CIN + c] = tobf(s);
}

__global__ void cast_bf16_kernel(const float* __restrict__ in, __bf16* __restrict__ out, int n) {
    int i = blockIdx.x * blockDim.x + threadIdx.x;
    if (i < n) out[i] = tobf(in[i]);
}

// ---------------------------------------------------------------------------
// 2) pos2embed + MLP + pack [x | pos] row as bf16 (ROWS x 288)
//    Faithfully reproduces the reference quirk: ez interleaves sin(pz), cos(px).
// ---------------------------------------------------------------------------
__global__ void posmlp_pack_kernel(const float* __restrict__ x,
                                   const float* __restrict__ coords,
                                   const float* __restrict__ pe_w1,
                                   const float* __restrict__ pe_b1,
                                   const float* __restrict__ pe_w2,
                                   const float* __restrict__ pe_b2,
                                   __bf16* __restrict__ out) {
    int row = blockIdx.x * blockDim.x + threadIdx.x;
    if (row >= ROWS) return;
    const float TWO_PI = 6.28318530717958647692f;
    float cx = coords[row * 4 + 1] * TWO_PI;
    float cy = coords[row * 4 + 2] * TWO_PI;
    float cz = coords[row * 4 + 3] * TWO_PI;
    float e[96];
#pragma unroll
    for (int j = 0; j < 16; ++j) {
        float dt = 1.0f + (float)j * 0.0625f;   // dim_t[2j] == dim_t[2j+1]
        float inv = 1.0f / dt;
        float ax = cx * inv, ay = cy * inv, az = cz * inv;
        e[2 * j]          = __sinf(ay); e[2 * j + 1]      = __cosf(ay);  // ey
        e[32 + 2 * j]     = __sinf(ax); e[32 + 2 * j + 1] = __cosf(ax);  // ex
        e[64 + 2 * j]     = __sinf(az); e[64 + 2 * j + 1] = __cosf(ax);  // ez (ref quirk)
    }
    float hdn[32];
#pragma unroll 4
    for (int o = 0; o < 32; ++o) {
        float s = pe_b1[o];
        const float* wr = pe_w1 + o * 96;
        for (int c = 0; c < 96; ++c) s += e[c] * wr[c];
        hdn[o] = fmaxf(s, 0.f);
    }
    __bf16* orow = out + (size_t)row * CIN;
    const float* xrow = x + (size_t)row * HIDC;
    for (int c = 0; c < HIDC; ++c) orow[c] = tobf(xrow[c]);
#pragma unroll 4
    for (int o = 0; o < 32; ++o) {
        float s = pe_b2[o];
        const float* wr = pe_w2 + o * 32;
        for (int c = 0; c < 32; ++c) s += hdn[c] * wr[c];
        orow[HIDC + o] = tobf(s);
    }
}

// ---------------------------------------------------------------------------
// 3) QKV GEMM:  [16384 x 288] @ eff^T -> q/k (bf16, row-major 16384x256),
//    v written transposed into vt (B,H,DH,N) bf16 for the P@V stage.
// ---------------------------------------------------------------------------
__global__ void gemm_qkv_kernel(const __bf16* __restrict__ qp,
                                const __bf16* __restrict__ ip,
                                const __bf16* __restrict__ eff,
                                const float* __restrict__ in_proj_b,
                                __bf16* __restrict__ qbf,
                                __bf16* __restrict__ kbf,
                                __bf16* __restrict__ vt) {
    const int wave = threadIdx.x >> 5, lane = threadIdx.x & 31;
    const int lrow = lane & 15, lhi = lane >> 4;
    const int kind = blockIdx.y;
    const int m0 = blockIdx.x * 16;
    const int n0w = wave * 64;

    const __bf16* A = (kind == 0) ? qp : ip;
    const __bf16* W = eff + (size_t)kind * HIDC * CIN;
    const float* bias = in_proj_b + kind * HIDC;

    v8f acc[4] = {v8f{}, v8f{}, v8f{}, v8f{}};
    const __bf16* arow = A + (size_t)(m0 + lrow) * CIN;

    for (int k0 = 0; k0 < CIN; k0 += 32) {
        AFrag a;
        int kb = k0 + (lhi << 4);
        a.h[0] = *(const v8bf*)(arow + kb);
        a.h[1] = *(const v8bf*)(arow + kb + 8);
#pragma unroll
        for (int t = 0; t < 4; ++t) {
            const __bf16* brow = W + (size_t)(n0w + t * 16 + lrow) * CIN + k0 + (lhi << 4);
            v16bf bv = *(const v16bf*)brow;
            acc[t] = wmma_bf16(a.v, bv, acc[t]);
        }
    }

    const int mb = m0 + (lhi << 3);
#pragma unroll
    for (int t = 0; t < 4; ++t) {
        int col = n0w + t * 16 + lrow;
        float bb = bias[col];
        if (kind < 2) {
            __bf16* dst = (kind == 0) ? qbf : kbf;
#pragma unroll
            for (int v = 0; v < 8; ++v)
                dst[(size_t)(mb + v) * HIDC + col] = tobf(acc[t][v] + bb);
        } else {
            int b = m0 / SEQN;
            int seq = (m0 % SEQN) + (lhi << 3);
            int h = col >> 6, d = col & 63;
            v8bf st;
#pragma unroll
            for (int v = 0; v < 8; ++v) st[v] = tobf(acc[t][v] + bb);
            *(v8bf*)(vt + ((size_t)(b * NHEADS + h) * DHEAD + d) * SEQN + seq) = st;
        }
    }
}

// ---------------------------------------------------------------------------
// 4) Flash attention per (b,h). Block = 256 thr (8 waves) = 128 query rows.
//    K/V tiles staged per 64-key block. TDM path: one tensor_load_to_lds DMA
//    per tile (double-buffered, TENSORcnt-pipelined, row padding de-conflicts
//    LDS banks). Fallback: cooperative register staging + XOR swizzle.
// ---------------------------------------------------------------------------
__global__ void attn_kernel(const __bf16* __restrict__ qbf,
                            const __bf16* __restrict__ kbf,
                            const __bf16* __restrict__ vt,
                            __bf16* __restrict__ obf) {
#if HAVE_TDM
    __shared__ __attribute__((aligned(64))) __bf16 kT[2][KBLK * KSTR + 16];
    __shared__ __attribute__((aligned(64))) __bf16 vTs[2][KBLK * KSTR + 16];
#else
    __shared__ __attribute__((aligned(64))) __bf16 kT[1][KBLK * KSTR];
    __shared__ __attribute__((aligned(64))) __bf16 vTs[1][KBLK * KSTR];
#endif
    __shared__ __attribute__((aligned(64))) __bf16 pst[8 * 16 * KBLK];

    const int tid = threadIdx.x;
    const int wave = tid >> 5, lane = tid & 31;
    const int lrow = lane & 15, lhi = lane >> 4;
    const int bh = blockIdx.y;
    const int b = bh >> 2, h = bh & 3;
    const int m0 = blockIdx.x * 128 + wave * 16;

    const __bf16* ksrc = kbf + (size_t)(b * SEQN) * HIDC + h * DHEAD;     // +key*HIDC
    const __bf16* vsrc = vt + (size_t)(b * NHEADS + h) * DHEAD * SEQN;    // +d*SEQN+key

    // q A-fragments (K = DH = 64 -> two 16x32 fragments), held for whole kernel
    const __bf16* qrow = qbf + ((size_t)(b * SEQN) + m0 + lrow) * HIDC + h * DHEAD;
    AFrag aq0, aq1;
    {
        int kb = lhi << 4;
        aq0.h[0] = *(const v8bf*)(qrow + kb);
        aq0.h[1] = *(const v8bf*)(qrow + kb + 8);
        aq1.h[0] = *(const v8bf*)(qrow + 32 + kb);
        aq1.h[1] = *(const v8bf*)(qrow + 32 + kb + 8);
    }

    v8f oacc[4] = {v8f{}, v8f{}, v8f{}, v8f{}};
    float mrun[8], lrun[8];
#pragma unroll
    for (int v = 0; v < 8; ++v) { mrun[v] = -INFINITY; lrun[v] = 0.f; }

#if !HAVE_TDM
    // cooperative staging: 64 rows x 128B per tile = 512 x 16B chunks
    v8bf kreg[2], vreg[2];
    auto prefetch = [&](int key0) {
#pragma unroll
        for (int i = 0; i < 2; ++i) {
            int c = tid + i * 256;
            int r = c >> 3, o = (c & 7) * 8;
            kreg[i] = *(const v8bf*)(ksrc + (size_t)(key0 + r) * HIDC + o);
            vreg[i] = *(const v8bf*)(vsrc + (size_t)r * SEQN + key0 + o);
        }
    };
    auto stage = [&]() {
#pragma unroll
        for (int i = 0; i < 2; ++i) {
            int c = tid + i * 256;
            int r = c >> 3, o = (c & 7) * 8;
            *(v8bf*)(kT[0] + r * KSTR + swz(r, o)) = kreg[i];
            *(v8bf*)(vTs[0] + r * KSTR + swz(r, o)) = vreg[i];
        }
    };
    prefetch(0);
#else
    if (wave == 0) {
        tdm_load_2d(ksrc, kT[0], HIDC);
        tdm_load_2d(vsrc, vTs[0], SEQN);
    }
#endif

    __bf16* pl = pst + wave * 16 * KBLK;

    for (int kb = 0; kb < SEQN / KBLK; ++kb) {
#if HAVE_TDM
        const int cur = kb & 1;
        if (wave == 0) __builtin_amdgcn_s_wait_tensorcnt(0);  // tile kb landed
        __syncthreads();   // broadcast completion; prev buffer fully consumed
        if (wave == 0 && kb + 1 < SEQN / KBLK) {
            tdm_load_2d(ksrc + (size_t)(kb + 1) * KBLK * HIDC, kT[cur ^ 1], HIDC);
            tdm_load_2d(vsrc + (kb + 1) * KBLK, vTs[cur ^ 1], SEQN);
        }
        const __bf16* kTile = kT[cur];
        const __bf16* vTile = vTs[cur];
#else
        __syncthreads();                 // previous tile fully consumed
        stage();
        __syncthreads();                 // tile visible to all waves
        if (kb + 1 < SEQN / KBLK) prefetch((kb + 1) * KBLK);  // fly during compute
        const __bf16* kTile = kT[0];
        const __bf16* vTile = vTs[0];
#endif

        // ---- S = q @ k^T : four 16x16 tiles over this 64-key block
        v8f s[4] = {v8f{}, v8f{}, v8f{}, v8f{}};
#pragma unroll
        for (int t = 0; t < 4; ++t) {
            int kr = t * 16 + lrow;
            const __bf16* krow = kTile + kr * KSTR;
            BFrag b0, b1;
#if HAVE_TDM
            b0.v = *(const v16bf*)(krow + (lhi << 4));
            b1.v = *(const v16bf*)(krow + (lhi << 4) + 32);
#else
            b0.h[0] = *(const v8bf*)(krow + swz(kr, (lhi << 4)));
            b0.h[1] = *(const v8bf*)(krow + swz(kr, (lhi << 4) + 8));
            b1.h[0] = *(const v8bf*)(krow + swz(kr, (lhi << 4) + 32));
            b1.h[1] = *(const v8bf*)(krow + swz(kr, (lhi << 4) + 40));
#endif
            s[t] = wmma_bf16(aq0.v, b0.v, s[t]);
            s[t] = wmma_bf16(aq1.v, b1.v, s[t]);
        }

        // ---- online softmax over 64 keys (rows live in a 16-lane half)
        float alpha[8];
#pragma unroll
        for (int v = 0; v < 8; ++v) {
            float a0 = s[0][v] * 0.125f, a1 = s[1][v] * 0.125f;
            float a2 = s[2][v] * 0.125f, a3 = s[3][v] * 0.125f;
            float rm = fmaxf(fmaxf(a0, a1), fmaxf(a2, a3));
            rm = fmaxf(rm, __shfl_xor(rm, 8, 32));
            rm = fmaxf(rm, __shfl_xor(rm, 4, 32));
            rm = fmaxf(rm, __shfl_xor(rm, 2, 32));
            rm = fmaxf(rm, __shfl_xor(rm, 1, 32));
            float mnew = fmaxf(mrun[v], rm);
            float p0 = __expf(a0 - mnew), p1 = __expf(a1 - mnew);
            float p2 = __expf(a2 - mnew), p3 = __expf(a3 - mnew);
            float sum = (p0 + p1) + (p2 + p3);
            sum += __shfl_xor(sum, 8, 32);
            sum += __shfl_xor(sum, 4, 32);
            sum += __shfl_xor(sum, 2, 32);
            sum += __shfl_xor(sum, 1, 32);
            alpha[v] = __expf(mrun[v] - mnew);
            lrun[v] = lrun[v] * alpha[v] + sum;
            mrun[v] = mnew;
            s[0][v] = p0; s[1][v] = p1; s[2][v] = p2; s[3][v] = p3;
        }
#pragma unroll
        for (int j = 0; j < 4; ++j)
#pragma unroll
            for (int v = 0; v < 8; ++v) oacc[j][v] *= alpha[v];

        // ---- P : C-layout -> A-layout via per-wave swizzled LDS tile (16x64)
#pragma unroll
        for (int t = 0; t < 4; ++t)
#pragma unroll
            for (int v = 0; v < 8; ++v) {
                int r = (lhi << 3) + v;
                pl[r * KBLK + swz(r, t * 16 + lrow)] = tobf(s[t][v]);
            }
        asm volatile("s_wait_dscnt 0" ::: "memory");
        AFrag pa0, pa1;
        {
            const __bf16* prow = pl + lrow * KBLK;
            int kb8 = lhi << 4;
            pa0.h[0] = *(const v8bf*)(prow + swz(lrow, kb8));
            pa0.h[1] = *(const v8bf*)(prow + swz(lrow, kb8 + 8));
            pa1.h[0] = *(const v8bf*)(prow + swz(lrow, kb8 + 32));
            pa1.h[1] = *(const v8bf*)(prow + swz(lrow, kb8 + 40));
        }

        // ---- O += P @ V  (V tile is [d][key]; contiguous along keys)
#pragma unroll
        for (int j = 0; j < 4; ++j) {
            int dr = j * 16 + lrow;
            const __bf16* vrow = vTile + dr * KSTR;
            BFrag b0, b1;
#if HAVE_TDM
            b0.v = *(const v16bf*)(vrow + (lhi << 4));
            b1.v = *(const v16bf*)(vrow + (lhi << 4) + 32);
#else
            b0.h[0] = *(const v8bf*)(vrow + swz(dr, (lhi << 4)));
            b0.h[1] = *(const v8bf*)(vrow + swz(dr, (lhi << 4) + 8));
            b1.h[0] = *(const v8bf*)(vrow + swz(dr, (lhi << 4) + 32));
            b1.h[1] = *(const v8bf*)(vrow + swz(dr, (lhi << 4) + 40));
#endif
            oacc[j] = wmma_bf16(pa0.v, b0.v, oacc[j]);
            oacc[j] = wmma_bf16(pa1.v, b1.v, oacc[j]);
        }
    }

    float inv[8];
#pragma unroll
    for (int v = 0; v < 8; ++v) inv[v] = 1.0f / lrun[v];
    const int mb = m0 + (lhi << 3);
#pragma unroll
    for (int j = 0; j < 4; ++j) {
        int col = h * DHEAD + j * 16 + lrow;
#pragma unroll
        for (int v = 0; v < 8; ++v)
            obf[((size_t)(b * SEQN) + mb + v) * HIDC + col] = tobf(oacc[j][v] * inv[v]);
    }
}

// ---------------------------------------------------------------------------
// 5) Out projection + bias + residual -> y (f32)
// ---------------------------------------------------------------------------
__global__ void gemm_out_kernel(const __bf16* __restrict__ obf,
                                const __bf16* __restrict__ wout,
                                const float* __restrict__ bias,
                                const float* __restrict__ Qin,
                                float* __restrict__ y) {
    const int wave = threadIdx.x >> 5, lane = threadIdx.x & 31;
    const int lrow = lane & 15, lhi = lane >> 4;
    const int m0 = blockIdx.x * 16;
    const int n0w = wave * 64;

    v8f acc[4] = {v8f{}, v8f{}, v8f{}, v8f{}};
    const __bf16* arow = obf + (size_t)(m0 + lrow) * HIDC;

    for (int k0 = 0; k0 < HIDC; k0 += 32) {
        AFrag a;
        int kb = k0 + (lhi << 4);
        a.h[0] = *(const v8bf*)(arow + kb);
        a.h[1] = *(const v8bf*)(arow + kb + 8);
#pragma unroll
        for (int t = 0; t < 4; ++t) {
            const __bf16* brow = wout + (size_t)(n0w + t * 16 + lrow) * HIDC + k0 + (lhi << 4);
            v16bf bv = *(const v16bf*)brow;
            acc[t] = wmma_bf16(a.v, bv, acc[t]);
        }
    }

    const int mb = m0 + (lhi << 3);
#pragma unroll
    for (int t = 0; t < 4; ++t) {
        int col = n0w + t * 16 + lrow;
        float bb = bias[col];
#pragma unroll
        for (int v = 0; v < 8; ++v) {
            size_t idx = (size_t)(mb + v) * HIDC + col;
            y[idx] = acc[t][v] + bb + Qin[idx];
        }
    }
}

// ---------------------------------------------------------------------------
// 6) LayerNorm: one wave per row (256 elems, 8 per lane)
// ---------------------------------------------------------------------------
__global__ void ln_kernel(const float* __restrict__ y,
                          const float* __restrict__ g,
                          const float* __restrict__ bb,
                          float* __restrict__ out) {
    const int wave = threadIdx.x >> 5, lane = threadIdx.x & 31;
    const int row = blockIdx.x * 8 + wave;
    const float* yr = y + (size_t)row * HIDC;
    float v[8];
    float s = 0.f;
#pragma unroll
    for (int i = 0; i < 8; ++i) { v[i] = yr[i * 32 + lane]; s += v[i]; }
    for (int off = 16; off >= 1; off >>= 1) s += __shfl_xor(s, off, 32);
    float mu = s * (1.0f / HIDC);
    float vs = 0.f;
#pragma unroll
    for (int i = 0; i < 8; ++i) { float d = v[i] - mu; vs += d * d; }
    for (int off = 16; off >= 1; off >>= 1) vs += __shfl_xor(vs, off, 32);
    float rstd = rsqrtf(vs * (1.0f / HIDC) + LN_EPS);
    float* orow = out + (size_t)row * HIDC;
#pragma unroll
    for (int i = 0; i < 8; ++i) {
        int c = i * 32 + lane;
        orow[c] = (v[i] - mu) * rstd * g[c] + bb[c];
    }
}

// ---------------------------------------------------------------------------
extern "C" void kernel_launch(void* const* d_in, const int* in_sizes, int n_in,
                              void* d_out, int out_size, void* d_ws, size_t ws_size,
                              hipStream_t stream) {
    const float* inputs       = (const float*)d_in[0];
    const float* Q_in         = (const float*)d_in[1];
    const float* input_coords = (const float*)d_in[2];
    const float* Q_in_coords  = (const float*)d_in[3];
    const float* Wq           = (const float*)d_in[4];
    const float* Wk           = (const float*)d_in[5];
    const float* Wv           = (const float*)d_in[6];
    const float* in_proj_w    = (const float*)d_in[7];
    const float* in_proj_b    = (const float*)d_in[8];
    const float* out_proj_w   = (const float*)d_in[9];
    const float* out_proj_b   = (const float*)d_in[10];
    const float* ln_g         = (const float*)d_in[11];
    const float* ln_b         = (const float*)d_in[12];
    const float* pe_w1        = (const float*)d_in[13];
    const float* pe_b1        = (const float*)d_in[14];
    const float* pe_w2        = (const float*)d_in[15];
    const float* pe_b2        = (const float*)d_in[16];

    // -------- workspace carve (y aliases the qp/ip region, dead by then) ----
    char* ws = (char*)d_ws;
    const size_t SZ_PACK = (size_t)ROWS * CIN * 2;        // 9,437,184
    const size_t SZ_MAT  = (size_t)ROWS * HIDC * 2;       // 8,388,608
    size_t off = 0;
    __bf16* qp   = (__bf16*)(ws + off);  off += SZ_PACK;
    __bf16* ipk  = (__bf16*)(ws + off);  off += SZ_PACK;
    float*  ybuf = (float*)(ws + 0);     // alias qp/ip region (16.8 MB < 18.9 MB)
    __bf16* eff  = (__bf16*)(ws + off);  off += (size_t)3 * HIDC * CIN * 2;
    __bf16* wout = (__bf16*)(ws + off);  off += (size_t)HIDC * HIDC * 2;
    __bf16* qbf  = (__bf16*)(ws + off);  off += SZ_MAT;
    __bf16* kbf  = (__bf16*)(ws + off);  off += SZ_MAT;
    __bf16* vt   = (__bf16*)(ws + off);  off += SZ_MAT;
    __bf16* obf  = (__bf16*)(ws + off);  off += SZ_MAT;
    (void)ws_size; (void)in_sizes; (void)n_in; (void)out_size;

    // 1) weight prep
    prep_eff_kernel<<<dim3((HIDC * CIN + 255) / 256, 3), 256, 0, stream>>>(
        in_proj_w, Wq, Wk, Wv, eff);
    cast_bf16_kernel<<<(HIDC * HIDC + 255) / 256, 256, 0, stream>>>(
        out_proj_w, wout, HIDC * HIDC);

    // 2) pos-MLP + pack (q-side and input-side)
    posmlp_pack_kernel<<<ROWS / 128, 128, 0, stream>>>(
        Q_in, Q_in_coords, pe_w1, pe_b1, pe_w2, pe_b2, qp);
    posmlp_pack_kernel<<<ROWS / 128, 128, 0, stream>>>(
        inputs, input_coords, pe_w1, pe_b1, pe_w2, pe_b2, ipk);

    // 3) fused QKV GEMM (y-dim selects q/k/v)
    gemm_qkv_kernel<<<dim3(ROWS / 16, 3), 128, 0, stream>>>(
        qp, ipk, eff, in_proj_b, qbf, kbf, vt);

    // 4) flash attention per (b,h), TDM/LDS-staged K/V tiles
    attn_kernel<<<dim3(SEQN / 128, BATCH * NHEADS), 256, 0, stream>>>(
        qbf, kbf, vt, obf);

    // 5) out projection + residual
    gemm_out_kernel<<<ROWS / 16, 128, 0, stream>>>(obf, wout, out_proj_b, Q_in, ybuf);

    // 6) layernorm -> d_out (f32)
    ln_kernel<<<ROWS / 8, 256, 0, stream>>>(ybuf, ln_g, ln_b, (float*)d_out);
}
